// GroupedQueryAttention_50233937494503
// MI455X (gfx1250) — matrix operand order repaired and checked by reading, not compiled, not run on hardware
//
#include <hip/hip_runtime.h>

typedef _Float16 v16h __attribute__((ext_vector_type(16)));
typedef _Float16 v8h  __attribute__((ext_vector_type(8)));
typedef _Float16 v4h  __attribute__((ext_vector_type(4)));
typedef float    v8f  __attribute__((ext_vector_type(8)));

#define S_LEN 4096
#define EMB   2048
#define KVD   512
#define HD    128
#define NHEAD 16

static __device__ __forceinline__ v8f wmma_f16(v16h a, v16h b, v8f c) {
    // D = A(16x32 f16) * B(32x16 f16) + C(16x16 f32)
    return __builtin_amdgcn_wmma_f32_16x16x32_f16(false, a, false, b, (short)0, c, false, false);
}

static __device__ __forceinline__ v16h cat8(v8h lo, v8h hi) {
    return __builtin_shufflevector(lo, hi, 0,1,2,3,4,5,6,7,8,9,10,11,12,13,14,15);
}

// ---------------- elementwise f32 -> f16 convert (x) ----------------
__global__ void __launch_bounds__(256) cvt_x(const float4* __restrict__ x, v4h* __restrict__ xh) {
    const size_t i = (size_t)blockIdx.x * blockDim.x + threadIdx.x;
    const float4 v = x[i];
    v4h o;
    o[0] = (_Float16)v.x; o[1] = (_Float16)v.y; o[2] = (_Float16)v.z; o[3] = (_Float16)v.w;
    xh[i] = o;
}

// ---------------- tiled transpose + convert: W[K][N] f32 -> WT[N][K] f16 ----------------
__global__ void __launch_bounds__(256) tr_cvt(const float* __restrict__ W, _Float16* __restrict__ WT,
                                              int K, int N) {
    __shared__ float tile[32][33];
    const int n0 = blockIdx.x * 32, k0 = blockIdx.y * 32;
    const int tx = threadIdx.x, ty = threadIdx.y;   // block (32,8)
    #pragma unroll
    for (int j = 0; j < 32; j += 8)
        tile[ty + j][tx] = W[(size_t)(k0 + ty + j) * N + n0 + tx];
    __syncthreads();
    #pragma unroll
    for (int j = 0; j < 32; j += 8)
        WT[(size_t)(n0 + ty + j) * K + k0 + tx] = (_Float16)tile[tx][ty + j];
}

// ---------------- WMMA GEMM: C[M][N] = A[M][K] * BT[N][K]^T + bias ----------------
// 4 waves/block; each wave computes a 16x64 tile. mode: 0=f16 store, 1=f16 transposed store, 2=f32 store
__global__ void __launch_bounds__(128) gemm_wmma(const _Float16* __restrict__ A,
                                                 const _Float16* __restrict__ BT,
                                                 const float* __restrict__ bias,
                                                 void* __restrict__ Cout,
                                                 int K, int ldc, float scale, int mode) {
    const int lane = threadIdx.x & 31;
    const int wv   = threadIdx.x >> 5;
    const int l16  = lane & 15;
    const int hl   = lane >> 4;
    const int m0 = (blockIdx.x * 4 + wv) * 16;
    const int n0 = blockIdx.y * 64;

    const v8f vzero = {0.f,0.f,0.f,0.f,0.f,0.f,0.f,0.f};
    v8f acc[4];
    #pragma unroll
    for (int t = 0; t < 4; ++t) acc[t] = vzero;

    for (int kc = 0; kc < K; kc += 32) {
        // A operand: lane<16 -> row m0+l16, K {kc..kc+7, kc+16..kc+23}; lane>=16 -> +8
        const _Float16* ap = A + (size_t)(m0 + l16) * K + kc + hl * 8;
        const v16h a = cat8(*(const v8h*)ap, *(const v8h*)(ap + 16));
        #pragma unroll
        for (int t = 0; t < 4; ++t) {
            // B operand: lane<16 -> col n0+16t+l16, K {kc..kc+7, kc+16..kc+23}; lane>=16 -> +8
            const _Float16* bp = BT + (size_t)(n0 + 16 * t + l16) * K + kc + hl * 8;
            acc[t] = wmma_f16(a, cat8(*(const v8h*)bp, *(const v8h*)(bp + 16)), acc[t]);
        }
    }
    #pragma unroll
    for (int t = 0; t < 4; ++t) {
        const int n = n0 + 16 * t + l16;
        const float bv = bias[n];
        if (mode == 1) {            // transposed f16 store: Cout[n][m], contiguous 8-half pack
            v8h o;
            #pragma unroll
            for (int r = 0; r < 8; ++r) o[r] = (_Float16)((acc[t][r] + bv) * scale);
            *(v8h*)((_Float16*)Cout + (size_t)n * ldc + m0 + 8 * hl) = o;
        } else if (mode == 2) {     // f32 store
            float* C = (float*)Cout;
            #pragma unroll
            for (int r = 0; r < 8; ++r)
                C[(size_t)(m0 + r + 8 * hl) * ldc + n] = (acc[t][r] + bv) * scale;
        } else {                    // f16 store
            _Float16* C = (_Float16*)Cout;
            #pragma unroll
            for (int r = 0; r < 8; ++r)
                C[(size_t)(m0 + r + 8 * hl) * ldc + n] = (_Float16)((acc[t][r] + bv) * scale);
        }
    }
}

// ---------------- fused causal GQA attention (flash-style, online softmax) ----------------
// One wave per 16-row Q tile. Qh pre-scaled by 1/sqrt(d). Kh row-major [S][512], Vt [512][S].
__global__ void __launch_bounds__(128) attn_kernel(const _Float16* __restrict__ Qh,
                                                   const _Float16* __restrict__ Kh,
                                                   const _Float16* __restrict__ Vt,
                                                   _Float16* __restrict__ ctxh) {
    __shared__ _Float16 ptile[4][16][32];   // per-wave 16x32 probability tile (f16)
    const int lane = threadIdx.x & 31;
    const int wv   = threadIdx.x >> 5;
    const int l16  = lane & 15;
    const int hl   = lane >> 4;
    const int h    = blockIdx.y;
    const int kvh  = h >> 2;                // 4 Q-heads share a KV head
    const int q0   = (blockIdx.x * 4 + wv) * 16;

    // Q tile 16x128 as four A operands (kept in registers)
    v16h aq[4];
    #pragma unroll
    for (int i = 0; i < 4; ++i) {
        const _Float16* ap = Qh + (size_t)(q0 + l16) * EMB + h * HD + 32 * i + hl * 8;
        aq[i] = cat8(*(const v8h*)ap, *(const v8h*)(ap + 16));
    }

    const v8f vzero = {0.f,0.f,0.f,0.f,0.f,0.f,0.f,0.f};
    v8f acc[8];                             // ctx accumulator 16x128
    #pragma unroll
    for (int dt = 0; dt < 8; ++dt) acc[dt] = vzero;
    float m[8], l[8];                       // per-row running max / sum (row = r + 8*hl)
    #pragma unroll
    for (int r = 0; r < 8; ++r) { m[r] = -__builtin_inff(); l[r] = 0.f; }

    for (int kb = 0; kb < q0 + 16; kb += 32) {
        // scores: two 16x16 tiles covering keys [kb,kb+16) and [kb+16,kb+32)
        v8f s0 = vzero, s1 = vzero;
        #pragma unroll
        for (int i = 0; i < 4; ++i) {
            const _Float16* kp0 = Kh + (size_t)(kb + l16)      * KVD + kvh * HD + 32 * i + hl * 8;
            const _Float16* kp1 = Kh + (size_t)(kb + 16 + l16) * KVD + kvh * HD + 32 * i + hl * 8;
            s0 = wmma_f16(aq[i], cat8(*(const v8h*)kp0, *(const v8h*)(kp0 + 16)), s0);
            s1 = wmma_f16(aq[i], cat8(*(const v8h*)kp1, *(const v8h*)(kp1 + 16)), s1);
        }
        // causal mask + tile row-max
        float t[8], alv[8], rs[8];
        #pragma unroll
        for (int r = 0; r < 8; ++r) {
            const int q = q0 + r + 8 * hl;
            if (kb + l16      > q) s0[r] = -__builtin_inff();
            if (kb + 16 + l16 > q) s1[r] = -__builtin_inff();
            t[r] = fmaxf(s0[r], s1[r]);
        }
        #pragma unroll
        for (int off = 1; off < 16; off <<= 1) {
            #pragma unroll
            for (int r = 0; r < 8; ++r)
                t[r] = fmaxf(t[r], __shfl_xor(t[r], off, 32));
        }
        // online softmax update + write P tile to LDS (D-layout -> memory)
        #pragma unroll
        for (int r = 0; r < 8; ++r) {
            const float mn = fmaxf(m[r], t[r]);
            alv[r] = __expf(m[r] - mn);
            m[r] = mn;
            const float p0 = __expf(s0[r] - mn);
            const float p1 = __expf(s1[r] - mn);
            rs[r] = p0 + p1;
            ptile[wv][r + 8 * hl][l16]      = (_Float16)p0;
            ptile[wv][r + 8 * hl][16 + l16] = (_Float16)p1;
        }
        #pragma unroll
        for (int off = 1; off < 16; off <<= 1) {
            #pragma unroll
            for (int r = 0; r < 8; ++r)
                rs[r] += __shfl_xor(rs[r], off, 32);
        }
        #pragma unroll
        for (int r = 0; r < 8; ++r) l[r] = l[r] * alv[r] + rs[r];
        #pragma unroll
        for (int dt = 0; dt < 8; ++dt) {
            #pragma unroll
            for (int r = 0; r < 8; ++r) acc[dt][r] *= alv[r];
        }
        // re-load P in A-operand layout (same-wave LDS ops are in-order)
        const _Float16* pp = &ptile[wv][l16][hl * 8];
        const v16h pa = cat8(*(const v8h*)pp, *(const v8h*)(pp + 16));
        // ctx += P(16x32) @ V(32x128): B operand rows come straight from Vt
        #pragma unroll
        for (int dt = 0; dt < 8; ++dt) {
            const _Float16* vp = Vt + (size_t)(kvh * HD + 16 * dt + l16) * S_LEN + kb + hl * 8;
            acc[dt] = wmma_f16(pa, cat8(*(const v8h*)vp, *(const v8h*)(vp + 16)), acc[dt]);
        }
    }
    // normalize + store ctx (f16, [S][EMB], head-major columns)
    #pragma unroll
    for (int dt = 0; dt < 8; ++dt) {
        #pragma unroll
        for (int r = 0; r < 8; ++r)
            ctxh[(size_t)(q0 + r + 8 * hl) * EMB + h * HD + 16 * dt + l16] =
                (_Float16)(acc[dt][r] / l[r]);
    }
}

extern "C" void kernel_launch(void* const* d_in, const int* in_sizes, int n_in,
                              void* d_out, int out_size, void* d_ws, size_t ws_size,
                              hipStream_t stream) {
    (void)in_sizes; (void)n_in; (void)out_size; (void)ws_size;
    const float* x    = (const float*)d_in[0];
    // d_in[1] = mask (causality computed analytically)
    const float* wq_w = (const float*)d_in[2];
    const float* wq_b = (const float*)d_in[3];
    const float* wk_w = (const float*)d_in[4];
    const float* wk_b = (const float*)d_in[5];
    const float* wv_w = (const float*)d_in[6];
    const float* wv_b = (const float*)d_in[7];
    const float* wo_w = (const float*)d_in[8];
    const float* wo_b = (const float*)d_in[9];

    _Float16* ws = (_Float16*)d_ws;
    _Float16* xh  = ws;                                  // [S][EMB]
    _Float16* Qh  = xh  + (size_t)S_LEN * EMB;           // [S][EMB] (pre-scaled)
    _Float16* Kh  = Qh  + (size_t)S_LEN * EMB;           // [S][KVD]
    _Float16* Vt  = Kh  + (size_t)S_LEN * KVD;           // [KVD][S] (transposed)
    _Float16* wqT = Vt  + (size_t)S_LEN * KVD;           // [EMB][EMB]
    _Float16* wkT = wqT + (size_t)EMB * EMB;             // [KVD][EMB]
    _Float16* wvT = wkT + (size_t)EMB * KVD;             // [KVD][EMB]
    _Float16* woT = wvT + (size_t)EMB * KVD;             // [EMB][EMB]
    _Float16* ctxh = xh;                                 // reuse: x dead after QKV

    // stage 1: convert / transpose to f16
    cvt_x<<<dim3((S_LEN * EMB / 4) / 256), 256, 0, stream>>>((const float4*)x, (v4h*)xh);
    tr_cvt<<<dim3(EMB / 32, EMB / 32), dim3(32, 8), 0, stream>>>(wq_w, wqT, EMB, EMB);
    tr_cvt<<<dim3(KVD / 32, EMB / 32), dim3(32, 8), 0, stream>>>(wk_w, wkT, EMB, KVD);
    tr_cvt<<<dim3(KVD / 32, EMB / 32), dim3(32, 8), 0, stream>>>(wv_w, wvT, EMB, KVD);
    tr_cvt<<<dim3(EMB / 32, EMB / 32), dim3(32, 8), 0, stream>>>(wo_w, woT, EMB, EMB);

    // stage 2: QKV projections (WMMA). Q pre-scaled by 1/sqrt(d); V stored transposed.
    const float qscale = 0.08838834764831845f;  // 128^-0.5
    gemm_wmma<<<dim3(S_LEN / 64, EMB / 64), 128, 0, stream>>>(xh, wqT, wq_b, Qh, EMB, EMB, qscale, 0);
    gemm_wmma<<<dim3(S_LEN / 64, KVD / 64), 128, 0, stream>>>(xh, wkT, wk_b, Kh, EMB, KVD, 1.0f, 0);
    gemm_wmma<<<dim3(S_LEN / 64, KVD / 64), 128, 0, stream>>>(xh, wvT, wv_b, Vt, EMB, S_LEN, 1.0f, 1);

    // stage 3: fused causal attention (flash-style, WMMA scores + WMMA ctx)
    attn_kernel<<<dim3(S_LEN / 64, NHEAD), 128, 0, stream>>>(Qh, Kh, Vt, ctxh);

    // stage 4: output projection, f32 store to d_out
    gemm_wmma<<<dim3(S_LEN / 64, EMB / 64), 128, 0, stream>>>(ctxh, woT, wo_b, d_out, EMB, EMB, 1.0f, 2);
}
